// BiLSTM_CRF_7404523618973
// MI455X (gfx1250) — compile-verified, matrix operand order checked by reading, not run on hardware
//
#include <hip/hip_runtime.h>
#include <hip/hip_bf16.h>
#include <cstdint>

// ---------------------------------------------------------------------------
// BiLSTM-CRF for MI455X (gfx1250, wave32, WMMA + TDM).
//  - All dense GEMMs (highway x6, LSTM input projections x4, tag proj) run on
//    v_wmma_f32_16x16x32_bf16 with fp32 accumulate.
//  - Recurrent LSTM: persistent 1-WG-per-direction kernel, 32 waves, h/c in
//    LDS, per-step barrier; 1024x256 matvec on the matrix pipe (h broadcast
//    as row 0 of the A fragment), Whh bf16 streamed from L2 each step.
//    Input-gate rows are double-buffered in LDS via TENSOR_LOAD_TO_LDS (TDM)
//    so the 4KB/step global read overlaps the matvec (s_wait_tensorcnt).
//  - CRF: re-associated as log-semiring matrix products -> chunk-parallel
//    scan (512 WGs x 32 steps) + one folding WG.
// ---------------------------------------------------------------------------

typedef __bf16 bf16_t;
typedef __attribute__((ext_vector_type(16))) __bf16 v16bf;
typedef __attribute__((ext_vector_type(8)))  float  v8f;
typedef __attribute__((ext_vector_type(4)))  unsigned int u32x4;
typedef __attribute__((ext_vector_type(8)))  int  i32x8;
typedef __attribute__((ext_vector_type(4)))  int  i32x4;

#define DEV __device__ __forceinline__

static constexpr int Dfull = 400;
static constexpr int Dpad  = 416;   // 13 * 32
static constexpr int Hh    = 256;
static constexpr int G4    = 1024;  // 4*H
static constexpr int H2    = 512;
static constexpr int Tpad  = 16;    // tagset 12 padded to 16
static constexpr int TSTART = 10, TSTOP = 11;
static constexpr float NEGV = -10000.0f;

DEV v8f vzero8() {
  v8f a;
#pragma unroll
  for (int i = 0; i < 8; ++i) a[i] = 0.0f;
  return a;
}

DEV bf16_t f2bf(float f) { return (bf16_t)f; }

// Load a 16x32 bf16 fragment (A layout per ISA 7.12.2):
//   lanes 0-15 : row = lane, K = base+0..7 (elems 0-7), base+16..23 (elems 8-15)
//   lanes 16-31: row = lane-16, K = base+8..15, base+24..31
// Two 16B loads per lane. Also used for the B operand of x*W^T (W row-major
// [N,K]): the per-lane row index then selects the weight row n = lane&15.
DEV v16bf load_frag(const bf16_t* __restrict__ p0, int ld) {
  int lane = threadIdx.x & 31;
  int r    = lane & 15;
  int koff = (lane & 16) ? 8 : 0;
  const bf16_t* p = p0 + (size_t)r * ld + koff;
  v16bf f;
  *reinterpret_cast<int4*>(&f)       = *reinterpret_cast<const int4*>(p);
  *(reinterpret_cast<int4*>(&f) + 1) = *reinterpret_cast<const int4*>(p + 16);
  return f;
}

DEV v8f wmma_bf16(v16bf a, v16bf b, v8f c) {
  return __builtin_amdgcn_wmma_f32_16x16x32_bf16(false, a, false, b, (short)0, c,
                                                 false, false);
}

DEV float sigmoidf_(float x) { return 1.0f / (1.0f + __expf(-x)); }

// ---------------------------------------------------------------------------
// TDM: DMA one contiguous row of `nelem` fp32 from global into LDS.
// D# bitfields per CDNA5 ISA:
//   group0: [1:0]=count=1, [63:32]=lds_addr, [120:64]=global_addr, [127:126]=2
//   group1: data_size=4B, tensor_dim0=nelem, tensor_dim1=1, tile_dim0=nelem,
//           tile_dim1=1, stride0=nelem; groups 2/3 zero (2-D descriptor).
// Issued by one wave only (TDM ignores EXEC); tracked with TENSORcnt.
// ---------------------------------------------------------------------------
DEV void tdm_load_row_f32(const float* gsrc, uint32_t lds_byte_off,
                          uint32_t nelem) {
  uint64_t ga = (uint64_t)(uintptr_t)gsrc;
  u32x4 g0;
  g0[0] = 1u;                                   // count=1 (valid user D#)
  g0[1] = lds_byte_off;                         // lds_addr (bytes)
  g0[2] = (uint32_t)(ga & 0xFFFFFFFFu);         // global_addr[31:0]
  g0[3] = (uint32_t)(ga >> 32) | (2u << 30);    // global_addr[56:32] | type=2
  i32x8 g1;
  g1[0] = (int)(2u << 16);                      // data_size = 4 bytes
  g1[1] = (int)((nelem & 0xFFFFu) << 16);       // tensor_dim0[15:0] in [31:16]
  g1[2] = (int)(((nelem >> 16) & 0xFFFFu) | (1u << 16));   // td0 hi | td1=1
  g1[3] = (int)((nelem & 0xFFFFu) << 16);       // td1 hi=0 | tile_dim0=nelem
  g1[4] = 1;                                    // tile_dim1=1, tile_dim2=0
  g1[5] = (int)nelem;                           // tensor_dim0_stride lo
  g1[6] = 0;
  g1[7] = 0;
  i32x4 z4;
#pragma unroll
  for (int q = 0; q < 4; ++q) z4[q] = 0;
#if __has_include(<hip/amd_detail/amd_gfx1250_TDM.h>)
  // therock-10.0 headers present -> clang-23 6-argument builtin
  i32x8 z8;
#pragma unroll
  for (int q = 0; q < 8; ++q) z8[q] = 0;
  __builtin_amdgcn_tensor_load_to_lds(g0, g1, z4, z4, z8, 0);
#else
  // ROCm 7.2 / clang-22 5-argument builtin
  __builtin_amdgcn_tensor_load_to_lds(g0, g1, z4, z4, 0);
#endif
}

// ---------------------------------------------------------------------------
// fp32 -> padded bf16 conversion (rows/cols beyond source are zero)
// ---------------------------------------------------------------------------
__global__ void cvt_pad_kernel(const float* __restrict__ src, bf16_t* __restrict__ dst,
                               int Rsrc, int Csrc, int Rdst, int Cdst) {
  size_t n = (size_t)Rdst * Cdst;
  for (size_t idx = blockIdx.x * (size_t)blockDim.x + threadIdx.x; idx < n;
       idx += (size_t)gridDim.x * blockDim.x) {
    int r = (int)(idx / Cdst);
    int c = (int)(idx % Cdst);
    float v = (r < Rsrc && c < Csrc) ? src[(size_t)r * Csrc + c] : 0.0f;
    dst[idx] = f2bf(v);
  }
}

__global__ void zero_tail_kernel(bf16_t* __restrict__ dst, int R, int c0, int Cdst) {
  int w = Cdst - c0;
  size_t n = (size_t)R * w;
  for (size_t idx = blockIdx.x * (size_t)blockDim.x + threadIdx.x; idx < n;
       idx += (size_t)gridDim.x * blockDim.x) {
    int r = (int)(idx / w);
    int c = (int)(idx % w);
    dst[(size_t)r * Cdst + c0 + c] = f2bf(0.0f);
  }
}

// ---------------------------------------------------------------------------
// Fused highway layer: y = sig(xWg^T+bg)*relu(xWn^T+bn)+(1-sig)*(xWl^T+bl)
// One wave per 16x16 output tile, three WMMA accumulators share the A frag.
// ---------------------------------------------------------------------------
__global__ __launch_bounds__(256) void highway_kernel(
    const bf16_t* __restrict__ X, const bf16_t* __restrict__ Wg,
    const bf16_t* __restrict__ Wn, const bf16_t* __restrict__ Wl,
    const float* __restrict__ bg, const float* __restrict__ bn,
    const float* __restrict__ bl, bf16_t* __restrict__ Y) {
  int wave = threadIdx.x >> 5, lane = threadIdx.x & 31;
  int m0 = blockIdx.x * 16;
  int ntile = blockIdx.y * 8 + wave;
  if (ntile * 16 >= Dfull) return;        // uniform per wave -> EXEC stays full
  int n0 = ntile * 16;

  v8f aG = vzero8(), aN = vzero8(), aL = vzero8();
  const bf16_t* Arow = X + (size_t)m0 * Dpad;
  for (int k0 = 0; k0 < Dpad; k0 += 32) {
    v16bf a  = load_frag(Arow + k0, Dpad);
    v16bf bG = load_frag(Wg + (size_t)n0 * Dpad + k0, Dpad);
    v16bf bN = load_frag(Wn + (size_t)n0 * Dpad + k0, Dpad);
    v16bf bL = load_frag(Wl + (size_t)n0 * Dpad + k0, Dpad);
    aG = wmma_bf16(a, bG, aG);
    aN = wmma_bf16(a, bN, aN);
    aL = wmma_bf16(a, bL, aL);
  }
  int n = n0 + (lane & 15);
  float bgv = bg[n], bnv = bn[n], blv = bl[n];
#pragma unroll
  for (int r = 0; r < 8; ++r) {
    int m = (lane < 16) ? r : (r + 8);
    float g  = sigmoidf_(aG[r] + bgv);
    float nv = aN[r] + bnv; nv = nv > 0.0f ? nv : 0.0f;
    float lv = aL[r] + blv;
    Y[(size_t)(m0 + m) * Dpad + n] = f2bf(g * nv + (1.0f - g) * lv);
  }
}

// ---------------------------------------------------------------------------
// Generic WMMA GEMM: C_f32[S,N] = A_bf16[S,Kpad] * W_bf16[N,Kpad]^T + bias
// Each wave owns 4 consecutive 16x16 n-tiles (A fragment reused x4).
// ---------------------------------------------------------------------------
__global__ __launch_bounds__(256) void gemm_bias_kernel(
    const bf16_t* __restrict__ A, int lda, const bf16_t* __restrict__ W, int ldw,
    const float* __restrict__ bias, int nbias, float* __restrict__ C, int ldc,
    int N, int K) {
  int wave = threadIdx.x >> 5, lane = threadIdx.x & 31;
  int m0  = blockIdx.x * 16;
  int nt0 = (blockIdx.y * 8 + wave) * 4;
  if (nt0 * 16 >= N) return;

  v8f acc[4];
#pragma unroll
  for (int t = 0; t < 4; ++t) acc[t] = vzero8();

  const bf16_t* Arow = A + (size_t)m0 * lda;
  for (int k0 = 0; k0 < K; k0 += 32) {
    v16bf a = load_frag(Arow + k0, lda);
#pragma unroll
    for (int t = 0; t < 4; ++t) {
      int n0 = (nt0 + t) * 16;
      if (n0 < N) {
        v16bf b = load_frag(W + (size_t)n0 * ldw + k0, ldw);
        acc[t] = wmma_bf16(a, b, acc[t]);
      }
    }
  }
#pragma unroll
  for (int t = 0; t < 4; ++t) {
    int n0 = (nt0 + t) * 16;
    if (n0 >= N) continue;
    int n = n0 + (lane & 15);
    float bv = (n < nbias) ? bias[n] : 0.0f;
#pragma unroll
    for (int r = 0; r < 8; ++r) {
      int m = (lane < 16) ? r : (r + 8);
      C[(size_t)(m0 + m) * ldc + n] = acc[t][r] + bv;
    }
  }
}

// ---------------------------------------------------------------------------
// Persistent LSTM scan: grid = 2 (dir 0 fwd, dir 1 bwd), 1024 threads (32
// waves). Per step:
//   1. wave 0 issues a TDM load of the NEXT step's input-gate row into the
//      other LDS buffer (overlaps with compute, tracked by TENSORcnt);
//   2. matvec g = h * Whh^T via WMMA (h as row 0 of the A fragment; each
//      wave produces 32 gate rows; result row 0 lives on lanes 0..15);
//   3. wave 0 waits tensorcnt, barrier, 256 threads do the cell update from
//      (wmma partial + DMA'd input gates), write h as bf16.
// ---------------------------------------------------------------------------
__global__ __launch_bounds__(1024) void lstm_scan_kernel(
    const float* __restrict__ gates_fwd, const float* __restrict__ gates_bwd,
    const bf16_t* __restrict__ whh_fwd, const bf16_t* __restrict__ whh_bwd,
    const float* __restrict__ h0, const float* __restrict__ c0, int hrow0,
    bf16_t* __restrict__ out, int out_ld, int S) {
  const int dir = blockIdx.x;  // 0 = forward, 1 = backward
  const float*  gates = dir ? gates_bwd : gates_fwd;
  const bf16_t* whh   = dir ? whh_bwd   : whh_fwd;
  const int hrow = hrow0 + dir;

  __shared__ float h_lds[Hh];
  __shared__ float c_lds[Hh];
  __shared__ float g_lds[G4];      // Whh*h partial sums
  __shared__ float gin[2][G4];     // TDM double buffer of input-gate rows

  int tid = threadIdx.x;
  int wave = tid >> 5, lane = tid & 31;
  if (tid < Hh) {
    h_lds[tid] = h0[(size_t)hrow * Hh + tid];
    c_lds[tid] = c0[(size_t)hrow * Hh + tid];
  }

  // Prologue: DMA step-0 gate row into buffer 0.
  if (wave == 0) {
    int s0 = dir ? (S - 1) : 0;
    tdm_load_row_f32(gates + (size_t)s0 * G4,
                     (uint32_t)(uintptr_t)&gin[0][0], G4);
  }
  __syncthreads();

  for (int t = 0; t < S; ++t) {
    int s = dir ? (S - 1 - t) : t;

    // (1) Kick off DMA for the next step's gate row into the other buffer.
    if (wave == 0 && t + 1 < S) {
      int sn = dir ? (S - 2 - t) : (t + 1);
      tdm_load_row_f32(gates + (size_t)sn * G4,
                       (uint32_t)(uintptr_t)&gin[(t + 1) & 1][0], G4);
    }

    // (2) Matvec on the matrix pipe. Wave `wave` -> gate rows [32w, 32w+32).
    v8f acc0 = vzero8(), acc1 = vzero8();
    int n0 = wave * 32;
    for (int k0 = 0; k0 < Hh; k0 += 32) {
      v16bf a;
#pragma unroll
      for (int e = 0; e < 16; ++e) a[e] = f2bf(0.0f);
      if ((lane & 15) == 0) {            // lanes 0 and 16 carry row 0 of A
        int kb = k0 + ((lane & 16) ? 8 : 0);
#pragma unroll
        for (int e = 0; e < 8; ++e) {
          a[e]     = f2bf(h_lds[kb + e]);
          a[8 + e] = f2bf(h_lds[kb + 16 + e]);
        }
      }
      v16bf b0 = load_frag(whh + (size_t)n0 * Hh + k0, Hh);
      v16bf b1 = load_frag(whh + (size_t)(n0 + 16) * Hh + k0, Hh);
      acc0 = wmma_bf16(a, b0, acc0);
      acc1 = wmma_bf16(a, b1, acc1);
    }
    if (lane < 16) {                     // D row 0: m=0 -> vgpr0, n = lane
      g_lds[n0 + lane]      = acc0[0];
      g_lds[n0 + 16 + lane] = acc1[0];
    }

    // (3) Current buffer's DMA must be complete (one newer TDM may be in
    // flight except on the last step), then make it visible to all waves.
    if (wave == 0) {
      if (t + 1 < S) __builtin_amdgcn_s_wait_tensorcnt(1);
      else           __builtin_amdgcn_s_wait_tensorcnt(0);
    }
    __syncthreads();

    const float* gi = gin[t & 1];
    if (tid < Hh) {
      float iv = sigmoidf_(g_lds[tid]          + gi[tid]);
      float fv = sigmoidf_(g_lds[Hh + tid]     + gi[Hh + tid]);
      float gg = tanhf   (g_lds[2 * Hh + tid] + gi[2 * Hh + tid]);
      float ov = sigmoidf_(g_lds[3 * Hh + tid] + gi[3 * Hh + tid]);
      float cn = fv * c_lds[tid] + iv * gg;
      c_lds[tid] = cn;
      float hn = ov * tanhf(cn);
      h_lds[tid] = hn;
      out[(size_t)s * out_ld + dir * Hh + tid] = f2bf(hn);
    }
    __syncthreads();                     // new h visible before next matvec
  }
}

// ---------------------------------------------------------------------------
// CRF as log-semiring matrix scan.  M_t[i,j] = trans[i,j] + feat_t[i];
// fv' = M_t (x) fv.  Composition is associative -> chunk-parallel.
// ---------------------------------------------------------------------------
__global__ __launch_bounds__(192) void crf_chunk_kernel(
    const float* __restrict__ feats, const float* __restrict__ trans,
    float* __restrict__ cm, int CH) {
  __shared__ float R[144], tr[144];
  int tid = threadIdx.x;
  if (tid < 144) tr[tid] = trans[tid];
  int t0 = blockIdx.x * CH;
  int i = tid / 12, j = tid % 12;
  if (tid < 144) R[tid] = tr[tid] + feats[(size_t)t0 * Tpad + i];
  __syncthreads();
  for (int t = t0 + 1; t < t0 + CH; ++t) {
    float val = 0.0f;
    if (tid < 144) {
      float ft = feats[(size_t)t * Tpad + i];
      float m = -1e30f;
#pragma unroll
      for (int k = 0; k < 12; ++k) m = fmaxf(m, tr[i * 12 + k] + R[k * 12 + j]);
      float ss = 0.0f;
#pragma unroll
      for (int k = 0; k < 12; ++k)
        ss += __expf(tr[i * 12 + k] + R[k * 12 + j] - m);
      val = ft + m + __logf(ss);
    }
    __syncthreads();
    if (tid < 144) R[tid] = val;
    __syncthreads();
  }
  if (tid < 144) cm[(size_t)blockIdx.x * 144 + tid] = R[tid];
}

__global__ __launch_bounds__(192) void crf_final_kernel(
    const float* __restrict__ cm, const float* __restrict__ trans,
    float* __restrict__ out, int nchunks) {
  __shared__ float Acc[144], tr[144], fv[12];
  int tid = threadIdx.x;
  if (tid < 144) { tr[tid] = trans[tid]; Acc[tid] = cm[tid]; }
  __syncthreads();
  int i = tid / 12, j = tid % 12;
  for (int c = 1; c < nchunks; ++c) {
    float val = 0.0f;
    if (tid < 144) {
      const float* M = cm + (size_t)c * 144;
      float m = -1e30f;
#pragma unroll
      for (int k = 0; k < 12; ++k) m = fmaxf(m, M[i * 12 + k] + Acc[k * 12 + j]);
      float ss = 0.0f;
#pragma unroll
      for (int k = 0; k < 12; ++k)
        ss += __expf(M[i * 12 + k] + Acc[k * 12 + j] - m);
      val = m + __logf(ss);
    }
    __syncthreads();
    if (tid < 144) Acc[tid] = val;
    __syncthreads();
  }
  if (tid < 12) {  // fv[i] = LSE_j(Acc[i,j] + init[j]) + trans[STOP,i]
    float m = -1e30f;
    for (int jj = 0; jj < 12; ++jj) {
      float ini = (jj == TSTART) ? 0.0f : NEGV;
      m = fmaxf(m, Acc[tid * 12 + jj] + ini);
    }
    float ss = 0.0f;
    for (int jj = 0; jj < 12; ++jj) {
      float ini = (jj == TSTART) ? 0.0f : NEGV;
      ss += __expf(Acc[tid * 12 + jj] + ini - m);
    }
    fv[tid] = m + __logf(ss) + tr[TSTOP * 12 + tid];
  }
  __syncthreads();
  if (tid == 0) {
    float m = -1e30f;
    for (int ii = 0; ii < 12; ++ii) m = fmaxf(m, fv[ii]);
    float ss = 0.0f;
    for (int ii = 0; ii < 12; ++ii) ss += __expf(fv[ii] - m);
    out[0] = m + __logf(ss);
  }
}

// ---------------------------------------------------------------------------
// Host-side orchestration
// ---------------------------------------------------------------------------
extern "C" void kernel_launch(void* const* d_in, const int* in_sizes, int n_in,
                              void* d_out, int out_size, void* d_ws, size_t ws_size,
                              hipStream_t stream) {
  const float* x     = (const float*)d_in[0];
  const float* Wg    = (const float*)d_in[1];
  const float* bg    = (const float*)d_in[2];
  const float* Wn    = (const float*)d_in[3];
  const float* bn    = (const float*)d_in[4];
  const float* Wl    = (const float*)d_in[5];
  const float* bl    = (const float*)d_in[6];
  const float* Wih0f = (const float*)d_in[7];
  const float* Whh0f = (const float*)d_in[8];
  const float* b0f   = (const float*)d_in[9];
  const float* Wih0b = (const float*)d_in[10];
  const float* Whh0b = (const float*)d_in[11];
  const float* b0b   = (const float*)d_in[12];
  const float* Wih1f = (const float*)d_in[13];
  const float* Whh1f = (const float*)d_in[14];
  const float* b1f   = (const float*)d_in[15];
  const float* Wih1b = (const float*)d_in[16];
  const float* Whh1b = (const float*)d_in[17];
  const float* b1b   = (const float*)d_in[18];
  const float* Wtag  = (const float*)d_in[19];
  const float* btag  = (const float*)d_in[20];
  const float* trans = (const float*)d_in[21];
  const float* h0    = (const float*)d_in[22];
  const float* c0    = (const float*)d_in[23];
  float* out = (float*)d_out;
  const int S = in_sizes[0] / Dfull;   // 16384

  // ---- workspace carve-out (256B aligned) ----
  uint8_t* base = (uint8_t*)d_ws;
  size_t off = 0;
  auto alloc = [&](size_t bytes) -> void* {
    off = (off + 255) & ~(size_t)255;
    void* p = base + off;
    off += bytes;
    return p;
  };
  bf16_t* xbA    = (bf16_t*)alloc((size_t)S * Dpad * 2);
  bf16_t* xbB    = (bf16_t*)alloc((size_t)S * Dpad * 2);
  bf16_t* x1     = (bf16_t*)alloc((size_t)S * H2 * 2);
  bf16_t* lo     = (bf16_t*)alloc((size_t)S * H2 * 2);
  float*  gatesF = (float*)alloc((size_t)S * G4 * 4);
  float*  gatesB = (float*)alloc((size_t)S * G4 * 4);
  float*  feats  = (float*)alloc((size_t)S * Tpad * 4);
  const int nch = 512;
  const int CH  = S / nch;
  float*  cmats = (float*)alloc((size_t)nch * 144 * 4);
  bf16_t* wGb    = (bf16_t*)alloc((size_t)2 * Dfull * Dpad * 2);
  bf16_t* wNb    = (bf16_t*)alloc((size_t)2 * Dfull * Dpad * 2);
  bf16_t* wLb    = (bf16_t*)alloc((size_t)2 * Dfull * Dpad * 2);
  bf16_t* wih0fb = (bf16_t*)alloc((size_t)G4 * Dpad * 2);
  bf16_t* wih0bb = (bf16_t*)alloc((size_t)G4 * Dpad * 2);
  bf16_t* whh0fb = (bf16_t*)alloc((size_t)G4 * Hh * 2);
  bf16_t* whh0bb = (bf16_t*)alloc((size_t)G4 * Hh * 2);
  bf16_t* whh1fb = (bf16_t*)alloc((size_t)G4 * Hh * 2);
  bf16_t* whh1bb = (bf16_t*)alloc((size_t)G4 * Hh * 2);
  bf16_t* wih1fb = (bf16_t*)alloc((size_t)G4 * H2 * 2);
  bf16_t* wih1bb = (bf16_t*)alloc((size_t)G4 * H2 * 2);
  bf16_t* wtagb  = (bf16_t*)alloc((size_t)Tpad * H2 * 2);

  auto cvt = [&](const float* s, bf16_t* d, int Rs, int Cs, int Rd, int Cd) {
    size_t n = (size_t)Rd * Cd;
    size_t b = (n + 255) / 256;
    unsigned blocks = (unsigned)(b > 4096 ? 4096 : b);
    cvt_pad_kernel<<<blocks, 256, 0, stream>>>(s, d, Rs, Cs, Rd, Cd);
  };

  // ---- stage 0: convert activations + weights to padded bf16 ----
  cvt(x, xbA, S, Dfull, S, Dpad);
  {
    size_t n = (size_t)S * (Dpad - Dfull);
    unsigned blocks = (unsigned)(((n + 255) / 256) > 4096 ? 4096 : ((n + 255) / 256));
    zero_tail_kernel<<<blocks, 256, 0, stream>>>(xbB, S, Dfull, Dpad);
  }
  cvt(Wg, wGb, 2 * Dfull, Dfull, 2 * Dfull, Dpad);
  cvt(Wn, wNb, 2 * Dfull, Dfull, 2 * Dfull, Dpad);
  cvt(Wl, wLb, 2 * Dfull, Dfull, 2 * Dfull, Dpad);
  cvt(Wih0f, wih0fb, G4, Dfull, G4, Dpad);
  cvt(Wih0b, wih0bb, G4, Dfull, G4, Dpad);
  cvt(Whh0f, whh0fb, G4, Hh, G4, Hh);
  cvt(Whh0b, whh0bb, G4, Hh, G4, Hh);
  cvt(Whh1f, whh1fb, G4, Hh, G4, Hh);
  cvt(Whh1b, whh1bb, G4, Hh, G4, Hh);
  cvt(Wih1f, wih1fb, G4, H2, G4, H2);
  cvt(Wih1b, wih1bb, G4, H2, G4, H2);
  cvt(Wtag, wtagb, 12, H2, Tpad, H2);

  // ---- stage 1: highway layers (ping-pong xbA <-> xbB) ----
  dim3 hwg(S / 16, 4);
  highway_kernel<<<hwg, 256, 0, stream>>>(xbA, wGb, wNb, wLb, bg, bn, bl, xbB);
  highway_kernel<<<hwg, 256, 0, stream>>>(
      xbB, wGb + (size_t)Dfull * Dpad, wNb + (size_t)Dfull * Dpad,
      wLb + (size_t)Dfull * Dpad, bg + Dfull, bn + Dfull, bl + Dfull, xbA);

  // ---- stage 2: layer-0 input projections + recurrent scan ----
  dim3 g0(S / 16, G4 / 512);
  gemm_bias_kernel<<<g0, 256, 0, stream>>>(xbA, Dpad, wih0fb, Dpad, b0f, G4,
                                           gatesF, G4, G4, Dpad);
  gemm_bias_kernel<<<g0, 256, 0, stream>>>(xbA, Dpad, wih0bb, Dpad, b0b, G4,
                                           gatesB, G4, G4, Dpad);
  lstm_scan_kernel<<<2, 1024, 0, stream>>>(gatesF, gatesB, whh0fb, whh0bb,
                                           h0, c0, 0, x1, H2, S);

  // ---- stage 3: layer-1 (reuse gate buffers) ----
  gemm_bias_kernel<<<g0, 256, 0, stream>>>(x1, H2, wih1fb, H2, b1f, G4,
                                           gatesF, G4, G4, H2);
  gemm_bias_kernel<<<g0, 256, 0, stream>>>(x1, H2, wih1bb, H2, b1b, G4,
                                           gatesB, G4, G4, H2);
  lstm_scan_kernel<<<2, 1024, 0, stream>>>(gatesF, gatesB, whh1fb, whh1bb,
                                           h0, c0, 2, lo, H2, S);

  // ---- stage 4: tag projection ----
  dim3 gt(S / 16, 1);
  gemm_bias_kernel<<<gt, 256, 0, stream>>>(lo, H2, wtagb, H2, btag, 12,
                                           feats, Tpad, Tpad, H2);

  // ---- stage 5: CRF log-partition (chunk-parallel log-semiring scan) ----
  crf_chunk_kernel<<<nch, 192, 0, stream>>>(feats, trans, cmats, CH);
  crf_final_kernel<<<1, 192, 0, stream>>>(cmats, trans, out, nch);
}
